// MultiHeadAttention_13159779795457
// MI455X (gfx1250) — matrix operand order repaired and checked by reading, not compiled, not run on hardware
//
#include <hip/hip_runtime.h>

// ---------------------------------------------------------------------------
// Multi-head attention forward for MI455X (gfx1250), wave32 + WMMA f16.
// out = concat( attn_output[2,2048,2048] f32 , attn_weights[2,16,2048,2048] f32 )
// ---------------------------------------------------------------------------

#define D_MODEL 2048
#define HEADS   16
#define HEAD_DIM 128
#define SEQ     2048
#define BATCH   2
#define TOKENS  (BATCH * SEQ)          // 4096

typedef __attribute__((ext_vector_type(16))) _Float16 v16h;
typedef __attribute__((ext_vector_type(4)))  _Float16 v4h;
typedef __attribute__((ext_vector_type(8)))  float    v8f;

// ---- WMMA fragment loaders (CDNA5 16x16x32 f16 layouts, wave32) -----------

// A fragment (16x32, MxK): lane L<16 holds row M=L, regs0-3 K=0..7, regs4-7 K=16..23;
// lane L>=16 holds row M=L-16, regs0-3 K=8..15, regs4-7 K=24..31.
__device__ __forceinline__ v16h ld_a16(const _Float16* base, int ld) {
  const int lane = threadIdx.x & 31;
  const int r = lane & 15, hi = lane >> 4;
  const _Float16* p = base + (size_t)r * ld + hi * 8;
  v16h f;
  uint4* u = reinterpret_cast<uint4*>(&f);
  u[0] = *reinterpret_cast<const uint4*>(p);        // 8 halves, K chunk 0
  u[1] = *reinterpret_cast<const uint4*>(p + 16);   // 8 halves, K chunk 1
  return f;
}

// B fragment (32x16, KxN) from row-major [N,K] memory (i.e. B^T): lane holds
// column N=lane&15; lanes 0-15 K=0..7,16..23, lanes 16-31 K=8..15,24..31.
__device__ __forceinline__ v16h ld_b16(const _Float16* base, int ld) {
  const int lane = threadIdx.x & 31;
  const int n = lane & 15, hi = lane >> 4;
  const _Float16* p = base + (size_t)n * ld + hi * 8;
  v16h f;
  uint4* u = reinterpret_cast<uint4*>(&f);
  u[0] = *reinterpret_cast<const uint4*>(p);
  u[1] = *reinterpret_cast<const uint4*>(p + 16);
  return f;
}

// A fragment converted on the fly from an f32 source (used for softmaxed P).
__device__ __forceinline__ v16h ld_a32(const float* base, int ld) {
  const int lane = threadIdx.x & 31;
  const int r = lane & 15, hi = lane >> 4;
  const float* p = base + (size_t)r * ld + hi * 8;
  float4 c0 = *reinterpret_cast<const float4*>(p);
  float4 c1 = *reinterpret_cast<const float4*>(p + 4);
  float4 c2 = *reinterpret_cast<const float4*>(p + 16);
  float4 c3 = *reinterpret_cast<const float4*>(p + 20);
  v16h f;
  f[0]=(_Float16)c0.x;  f[1]=(_Float16)c0.y;  f[2]=(_Float16)c0.z;  f[3]=(_Float16)c0.w;
  f[4]=(_Float16)c1.x;  f[5]=(_Float16)c1.y;  f[6]=(_Float16)c1.z;  f[7]=(_Float16)c1.w;
  f[8]=(_Float16)c2.x;  f[9]=(_Float16)c2.y;  f[10]=(_Float16)c2.z; f[11]=(_Float16)c2.w;
  f[12]=(_Float16)c3.x; f[13]=(_Float16)c3.y; f[14]=(_Float16)c3.z; f[15]=(_Float16)c3.w;
  return f;
}

__device__ __forceinline__ v8f wmma32(v16h a, v16h b, v8f c) {
  return __builtin_amdgcn_wmma_f32_16x16x32_f16(false, a, false, b, (short)0, c,
                                                false, false);
}

// ---------------------------------------------------------------------------
// f32 -> f16 bulk convert (float4-wide)
// ---------------------------------------------------------------------------
__global__ void __launch_bounds__(256)
k_cvt(const float* __restrict__ src, _Float16* __restrict__ dst, int n4) {
  int i = blockIdx.x * 256 + threadIdx.x;
  if (i < n4) {
    float4 v = reinterpret_cast<const float4*>(src)[i];
    v4h o;
    o[0] = (_Float16)v.x; o[1] = (_Float16)v.y;
    o[2] = (_Float16)v.z; o[3] = (_Float16)v.w;
    reinterpret_cast<v4h*>(dst)[i] = o;
  }
}

// ---------------------------------------------------------------------------
// Projection GEMM: Y[m,n] = sum_k A[m,k]*B[n,k] (+bias[n]); output f16.
// transpose_out: write Y as Vt[b][n][s] with m = b*SEQ + s (for V).
// One wave computes a 64x64 tile; 8 waves per block.
// ---------------------------------------------------------------------------
__global__ void __launch_bounds__(256)
k_gemm_qkv(const _Float16* __restrict__ A, const _Float16* __restrict__ B,
           const float* __restrict__ bias, _Float16* __restrict__ Y,
           int M, int N, int K, int transpose_out) {
  const int wave = threadIdx.x >> 5;
  const int lane = threadIdx.x & 31;
  const int tiles_n = N >> 6;
  const int t  = blockIdx.x * 8 + wave;
  const int M0 = (t / tiles_n) << 6;
  const int N0 = (t % tiles_n) << 6;

  v8f acc[4][4] = {};
  for (int kk = 0; kk < K; kk += 32) {
    v16h af[4], bf[4];
#pragma unroll
    for (int i = 0; i < 4; ++i) af[i] = ld_a16(A + (size_t)(M0 + 16*i) * K + kk, K);
#pragma unroll
    for (int j = 0; j < 4; ++j) bf[j] = ld_b16(B + (size_t)(N0 + 16*j) * K + kk, K);
#pragma unroll
    for (int i = 0; i < 4; ++i)
#pragma unroll
      for (int j = 0; j < 4; ++j) acc[i][j] = wmma32(af[i], bf[j], acc[i][j]);
  }

  const int cn = lane & 15, hi = lane >> 4;
#pragma unroll
  for (int i = 0; i < 4; ++i)
#pragma unroll
    for (int j = 0; j < 4; ++j) {
      const int n = N0 + 16*j + cn;
      const float bb = bias ? bias[n] : 0.0f;
#pragma unroll
      for (int r = 0; r < 8; ++r) {
        const int m = M0 + 16*i + r + 8*hi;
        const float val = acc[i][j][r] + bb;
        if (!transpose_out) {
          Y[(size_t)m * N + n] = (_Float16)val;
        } else {
          const int b = m >> 11, s = m & (SEQ - 1);
          Y[(((size_t)(b * D_MODEL + n)) << 11) + s] = (_Float16)val;
        }
      }
    }
}

// ---------------------------------------------------------------------------
// S = scale * Q Kh^T per (batch,head); M=N=2048, K=128; f32 out to d_out.
// grid = (128, 32), block = 256 (8 waves of 64x64 tiles).
// ---------------------------------------------------------------------------
__global__ void __launch_bounds__(256)
k_gemm_qk(const _Float16* __restrict__ Qh, const _Float16* __restrict__ Kh,
          float* __restrict__ Sout, float scale) {
  const int wave = threadIdx.x >> 5;
  const int lane = threadIdx.x & 31;
  const int z = blockIdx.y;            // b*16 + h
  const int b = z >> 4, h = z & 15;
  const int t  = blockIdx.x * 8 + wave;   // 0..1023 (32x32 tiles)
  const int M0 = (t >> 5) << 6;
  const int N0 = (t & 31) << 6;

  const _Float16* Ab = Qh + ((size_t)b << 22) + h * HEAD_DIM;
  const _Float16* Bb = Kh + ((size_t)b << 22) + h * HEAD_DIM;
  float* S = Sout + ((size_t)z << 22);

  v8f acc[4][4] = {};
#pragma unroll
  for (int kk = 0; kk < HEAD_DIM; kk += 32) {
    v16h af[4], bf[4];
#pragma unroll
    for (int i = 0; i < 4; ++i) af[i] = ld_a16(Ab + ((size_t)(M0 + 16*i) << 11) + kk, D_MODEL);
#pragma unroll
    for (int j = 0; j < 4; ++j) bf[j] = ld_b16(Bb + ((size_t)(N0 + 16*j) << 11) + kk, D_MODEL);
#pragma unroll
    for (int i = 0; i < 4; ++i)
#pragma unroll
      for (int j = 0; j < 4; ++j) acc[i][j] = wmma32(af[i], bf[j], acc[i][j]);
  }

  const int cn = lane & 15, hi = lane >> 4;
#pragma unroll
  for (int i = 0; i < 4; ++i)
#pragma unroll
    for (int j = 0; j < 4; ++j) {
      const int n = N0 + 16*j + cn;
#pragma unroll
      for (int r = 0; r < 8; ++r) {
        const int m = M0 + 16*i + r + 8*hi;
        S[((size_t)m << 11) + n] = acc[i][j][r] * scale;
      }
    }
}

// ---------------------------------------------------------------------------
// Row softmax in place over 2048-wide rows. One 256-thread block per row.
// ---------------------------------------------------------------------------
__global__ void __launch_bounds__(256)
k_softmax(float* __restrict__ W) {
  __shared__ float red[256];
  float* p = W + ((size_t)blockIdx.x << 11);
  const int t = threadIdx.x;

  float vals[8];
  float m = -3.0e38f;
#pragma unroll
  for (int i = 0; i < 8; ++i) { vals[i] = p[t + 256 * i]; m = fmaxf(m, vals[i]); }
  red[t] = m; __syncthreads();
  for (int s = 128; s > 0; s >>= 1) {
    if (t < s) red[t] = fmaxf(red[t], red[t + s]);
    __syncthreads();
  }
  m = red[0]; __syncthreads();

  float sum = 0.0f;
#pragma unroll
  for (int i = 0; i < 8; ++i) { vals[i] = __expf(vals[i] - m); sum += vals[i]; }
  red[t] = sum; __syncthreads();
  for (int s = 128; s > 0; s >>= 1) {
    if (t < s) red[t] += red[t + s];
    __syncthreads();
  }
  const float inv = 1.0f / red[0];
#pragma unroll
  for (int i = 0; i < 8; ++i) p[t + 256 * i] = vals[i] * inv;
}

// ---------------------------------------------------------------------------
// C[b,q,h*128+d] = sum_k P[z,q,k] * V[b,k,h*128+d], via Vt[b][n][s] f16.
// A read from f32 softmax output (converted per fragment). grid=(8,32).
// ---------------------------------------------------------------------------
__global__ void __launch_bounds__(256)
k_gemm_pv(const float* __restrict__ P, const _Float16* __restrict__ Vt,
          _Float16* __restrict__ Ch) {
  const int wave = threadIdx.x >> 5;
  const int lane = threadIdx.x & 31;
  const int z = blockIdx.y;
  const int b = z >> 4, h = z & 15;
  const int t  = blockIdx.x * 8 + wave;   // 0..63 (32 x 2 tiles)
  const int M0 = (t >> 1) << 6;
  const int N0 = (t & 1) << 6;

  const float*    Pz = P  + ((size_t)z << 22);
  const _Float16* Bb = Vt + ((size_t)(b * D_MODEL + h * HEAD_DIM) << 11);

  v8f acc[4][4] = {};
  for (int kk = 0; kk < SEQ; kk += 32) {
    v16h af[4], bf[4];
#pragma unroll
    for (int i = 0; i < 4; ++i) af[i] = ld_a32(Pz + ((size_t)(M0 + 16*i) << 11) + kk, SEQ);
#pragma unroll
    for (int j = 0; j < 4; ++j) bf[j] = ld_b16(Bb + ((size_t)(N0 + 16*j) << 11) + kk, SEQ);
#pragma unroll
    for (int i = 0; i < 4; ++i)
#pragma unroll
      for (int j = 0; j < 4; ++j) acc[i][j] = wmma32(af[i], bf[j], acc[i][j]);
  }

  const int cn = lane & 15, hi = lane >> 4;
#pragma unroll
  for (int i = 0; i < 4; ++i)
#pragma unroll
    for (int j = 0; j < 4; ++j) {
      const int n = N0 + 16*j + cn;
#pragma unroll
      for (int r = 0; r < 8; ++r) {
        const int m = M0 + 16*i + r + 8*hi;
        Ch[(((size_t)(b * SEQ + m)) << 11) + h * HEAD_DIM + n] = (_Float16)acc[i][j][r];
      }
    }
}

// ---------------------------------------------------------------------------
// out[m,n] = sum_k Ch[m,k]*Woh[n,k] + bo[n]; f32 out. M=4096, N=K=2048.
// ---------------------------------------------------------------------------
__global__ void __launch_bounds__(256)
k_gemm_out(const _Float16* __restrict__ A, const _Float16* __restrict__ B,
           const float* __restrict__ bias, float* __restrict__ Y) {
  const int wave = threadIdx.x >> 5;
  const int lane = threadIdx.x & 31;
  const int t  = blockIdx.x * 8 + wave;   // 0..2047 (64 x 32 tiles)
  const int M0 = (t >> 5) << 6;
  const int N0 = (t & 31) << 6;

  v8f acc[4][4] = {};
  for (int kk = 0; kk < D_MODEL; kk += 32) {
    v16h af[4], bf[4];
#pragma unroll
    for (int i = 0; i < 4; ++i) af[i] = ld_a16(A + ((size_t)(M0 + 16*i) << 11) + kk, D_MODEL);
#pragma unroll
    for (int j = 0; j < 4; ++j) bf[j] = ld_b16(B + ((size_t)(N0 + 16*j) << 11) + kk, D_MODEL);
#pragma unroll
    for (int i = 0; i < 4; ++i)
#pragma unroll
      for (int j = 0; j < 4; ++j) acc[i][j] = wmma32(af[i], bf[j], acc[i][j]);
  }

  const int cn = lane & 15, hi = lane >> 4;
#pragma unroll
  for (int i = 0; i < 4; ++i)
#pragma unroll
    for (int j = 0; j < 4; ++j) {
      const int n = N0 + 16*j + cn;
      const float bb = bias[n];
#pragma unroll
      for (int r = 0; r < 8; ++r) {
        const int m = M0 + 16*i + r + 8*hi;
        Y[((size_t)m << 11) + n] = acc[i][j][r] + bb;
      }
    }
}

// ---------------------------------------------------------------------------
// Host-side orchestration
// ---------------------------------------------------------------------------
extern "C" void kernel_launch(void* const* d_in, const int* in_sizes, int n_in,
                              void* d_out, int out_size, void* d_ws, size_t ws_size,
                              hipStream_t stream) {
  (void)in_sizes; (void)n_in; (void)out_size; (void)ws_size;

  const float* X  = (const float*)d_in[0];   // [2,2048,2048]
  const float* Wq = (const float*)d_in[1];
  const float* bq = (const float*)d_in[2];
  const float* Wk = (const float*)d_in[3];
  const float* Wv = (const float*)d_in[4];
  const float* bv = (const float*)d_in[5];
  const float* Wo = (const float*)d_in[6];
  const float* bo = (const float*)d_in[7];

  float* out   = (float*)d_out;                               // attn_output
  float* attnw = out + (size_t)BATCH * SEQ * D_MODEL;         // attn_weights

  // Workspace layout (f16), total ~112 MiB.
  const size_t NX = (size_t)TOKENS * D_MODEL;     // 8,388,608
  const size_t NW = (size_t)D_MODEL * D_MODEL;    // 4,194,304
  _Float16* Xh  = (_Float16*)d_ws;
  _Float16* Wqh = Xh  + NX;
  _Float16* Wkh = Wqh + NW;
  _Float16* Wvh = Wkh + NW;
  _Float16* Woh = Wvh + NW;
  _Float16* Qh  = Woh + NW;
  _Float16* Kh  = Qh  + NX;
  _Float16* Vt  = Kh  + NX;   // [2][d_model n][seq s]
  _Float16* Ch  = Vt  + NX;

  // 1) f32 -> f16 converts
  k_cvt<<<(int)(NX / 4 / 256), 256, 0, stream>>>(X,  Xh,  (int)(NX / 4));
  k_cvt<<<(int)(NW / 4 / 256), 256, 0, stream>>>(Wq, Wqh, (int)(NW / 4));
  k_cvt<<<(int)(NW / 4 / 256), 256, 0, stream>>>(Wk, Wkh, (int)(NW / 4));
  k_cvt<<<(int)(NW / 4 / 256), 256, 0, stream>>>(Wv, Wvh, (int)(NW / 4));
  k_cvt<<<(int)(NW / 4 / 256), 256, 0, stream>>>(Wo, Woh, (int)(NW / 4));

  // 2) Q/K/V projections (64x64 tile per wave; 2048 tiles -> 256 blocks)
  k_gemm_qkv<<<256, 256, 0, stream>>>(Xh, Wqh, bq,      Qh, TOKENS, D_MODEL, D_MODEL, 0);
  k_gemm_qkv<<<256, 256, 0, stream>>>(Xh, Wkh, nullptr, Kh, TOKENS, D_MODEL, D_MODEL, 0);
  k_gemm_qkv<<<256, 256, 0, stream>>>(Xh, Wvh, bv,      Vt, TOKENS, D_MODEL, D_MODEL, 1);

  // 3) S = scale * Q K^T  -> attn_weights region of d_out
  const float scale = 0.08838834764831845f;   // 1/sqrt(128)
  k_gemm_qk<<<dim3(128, BATCH * HEADS), 256, 0, stream>>>(Qh, Kh, attnw, scale);

  // 4) softmax rows in place (65536 rows of 2048)
  k_softmax<<<BATCH * HEADS * SEQ, 256, 0, stream>>>(attnw);

  // 5) context = P @ V (per batch/head), f16 context
  k_gemm_pv<<<dim3(8, BATCH * HEADS), 256, 0, stream>>>(attnw, Vt, Ch);

  // 6) output projection + bias -> attn_output region of d_out
  k_gemm_out<<<256, 256, 0, stream>>>(Ch, Woh, bo, out);
}
